// SCOT_69844758167643
// MI455X (gfx1250) — matrix operand order repaired and checked by reading, not compiled, not run on hardware
//
#include <hip/hip_runtime.h>
#include <hip/hip_bf16.h>

typedef __attribute__((ext_vector_type(16))) __bf16 v16bf;
typedef __attribute__((ext_vector_type(8)))  __bf16 v8bf;
typedef __attribute__((ext_vector_type(8)))  float  v8f;

#define NCELLS 256
#define NGENES 2000
// leaky_relu(x) == max(x, 0.01x): 2 VALU ops instead of mul+cmp+cndmask
#define LRELU(x) fmaxf((x), 0.01f * (x))

// padded LDS leading dims (elements): +4 dwords/row breaks the 64-bank
// alignment (row stride 528B = 33*16B -> rows shift 4 banks each)
#define HLD 264   // for 256-wide h
#define QLD 136   // for 128-wide activations

__device__ __forceinline__ v8f wmma_bf16(v16bf a, v16bf b, v8f c) {
  return __builtin_amdgcn_wmma_f32_16x16x32_bf16(false, a, false, b, (short)0, c,
                                                 false, false);
}

// A-operand 16x32 bf16 fragment from row-major LDS tile.
// ISA 7.12.2: lanes 0-15 hold M=lane, K={0..7,16..23}; lanes 16-31 hold
// M=lane-16, K={8..15,24..31}: two contiguous 16B runs -> 2x ds_load_b128.
__device__ __forceinline__ v16bf load_a_lds(const __bf16* buf, int ld, int row,
                                            int kbase, int half) {
  const v8bf lo = *(const v8bf*)(buf + row * ld + kbase + half * 8);
  const v8bf hi = *(const v8bf*)(buf + row * ld + kbase + 16 + half * 8);
  v16bf a;
#pragma unroll
  for (int i = 0; i < 8; ++i) { a[i] = lo[i]; a[i + 8] = hi[i]; }
  return a;
}

// ---------------- prep kernels ----------------

__global__ void mlp_layer(const float* __restrict__ x, const float* __restrict__ W,
                          const float* __restrict__ b, float* __restrict__ y,
                          int M, int K, int Nc, int relu) {
  int t = blockIdx.x * blockDim.x + threadIdx.x;
  if (t >= M * Nc) return;
  int m = t / Nc, n = t % Nc;
  float s = b[n];
  for (int k = 0; k < K; ++k) s += x[(size_t)m * K + k] * W[(size_t)k * Nc + n];
  if (relu) s = LRELU(s);
  y[t] = s;
}

// B operand (KxN row-major f32) -> per-lane contiguous WMMA fragments:
// frag(kt*Ntiles+nt), lane, elem i = B[kt*32 + (i<8 ? half*8+i : 16+half*8+i-8)][nt*16 + (lane&15)]
__global__ void swizzle_b(const float* __restrict__ src, __bf16* __restrict__ dst,
                          int Krows, int Ncols) {
  int Ktiles = Krows >> 5, Ntiles = Ncols >> 4;
  int t = blockIdx.x * blockDim.x + threadIdx.x;
  if (t >= Ktiles * Ntiles * 32) return;
  int lane = t & 31, frag = t >> 5;
  int nt = frag % Ntiles, kt = frag / Ntiles;
  int half = lane >> 4, n = nt * 16 + (lane & 15);
  __bf16* out = dst + ((size_t)frag * 32 + lane) * 16;
#pragma unroll
  for (int i = 0; i < 16; ++i) {
    int k = kt * 32 + ((i < 8) ? (half * 8 + i) : (16 + half * 8 + (i - 8)));
    out[i] = (__bf16)src[(size_t)k * Ncols + n];
  }
}

// giW reordered to A-order [ce(64) | ctrl(1) | zero-pad to 96] x 256
__global__ void swizzle_giw(const float* __restrict__ giW, __bf16* __restrict__ dst) {
  int t = blockIdx.x * blockDim.x + threadIdx.x;  // 3 kt * 16 nt * 32 lanes
  if (t >= 3 * 16 * 32) return;
  int lane = t & 31, frag = t >> 5;
  int nt = frag % 16, kt = frag / 16;
  int half = lane >> 4, n = nt * 16 + (lane & 15);
  __bf16* out = dst + ((size_t)frag * 32 + lane) * 16;
#pragma unroll
  for (int i = 0; i < 16; ++i) {
    int k = kt * 32 + ((i < 8) ? (half * 8 + i) : (16 + half * 8 + (i - 8)));
    float v = 0.f;
    if (k < 64)       v = giW[(size_t)(1 + k) * 256 + n];  // ce rows
    else if (k == 64) v = giW[n];                          // ctrl row
    out[i] = (__bf16)v;
  }
}

// cell-embedding (256x64 f32) -> pre-built A fragments, frag(rt*2+kt)
__global__ void build_ce_frags(const float* __restrict__ ce, __bf16* __restrict__ dst) {
  int t = blockIdx.x * blockDim.x + threadIdx.x;  // 16 rt * 2 kt * 32 lanes
  if (t >= 16 * 2 * 32) return;
  int lane = t & 31, frag = t >> 5;
  int kt = frag & 1, rt = frag >> 1;
  int half = lane >> 4, row = rt * 16 + (lane & 15);
  __bf16* out = dst + ((size_t)frag * 32 + lane) * 16;
#pragma unroll
  for (int i = 0; i < 16; ++i) {
    int kl = (i < 8) ? (half * 8 + i) : (16 + half * 8 + (i - 8));
    out[i] = (__bf16)ce[(size_t)row * 64 + kt * 32 + kl];
  }
}

// ---------------- main fused kernel: one workgroup per gene ----------------

__global__ __launch_bounds__(256, 1)
void fcgnn_gene_kernel(const float* __restrict__ ctrl,
                       const float* __restrict__ mean_shift,
                       const float* __restrict__ gene_table,
                       const float* __restrict__ giW, const float* __restrict__ gib,
                       const float* __restrict__ m1aW, const float* __restrict__ m1ab,
                       const float* __restrict__ m1bb,
                       const float* __restrict__ m2aW, const float* __restrict__ m2ab,
                       const float* __restrict__ m2bb,
                       const float* __restrict__ oW, const float* __restrict__ ob,
                       const __bf16* __restrict__ ceFrag,
                       const __bf16* __restrict__ giWs,
                       const __bf16* __restrict__ m1aWs,
                       const __bf16* __restrict__ m1bWs,
                       const __bf16* __restrict__ m2aWs,
                       const __bf16* __restrict__ m2bWs,
                       float* __restrict__ out) {
  __shared__ __bf16 hbuf[NCELLS * HLD];   // ~132 KB: h (ld=HLD), then h2/h3 (ld=QLD)
  __shared__ __bf16 abuf[NCELLS * QLD];   //  ~68 KB: 'a' activations (ld=QLD)
  __shared__ float  crow[256];
  __shared__ float  pooled[256];
  __shared__ float  pvec[128];
  __shared__ __bf16 ctrlcol[NCELLS];
  __shared__ float  redm[8], reds[8];

  const int g = blockIdx.x;
  const int tid = threadIdx.x;
  const int lane = tid & 31;
  const int half = lane >> 4, l16 = lane & 15;
  // force wave index scalar: job loops become SALU-controlled, EXEC stays all-1s
  const int wv = __builtin_amdgcn_readfirstlane(tid >> 5);

  // per-gene setup: ctrl column; fold gene_embed/shift/N rows into const row
  ctrlcol[tid] = (__bf16)ctrl[(size_t)tid * NGENES + g];
  {
    int n = tid;
    float s = gib[n];
#pragma unroll 4
    for (int j = 0; j < 64; ++j)
      s += gene_table[(size_t)g * 64 + j] * giW[(size_t)(65 + j) * 256 + n];
    s += mean_shift[g] * giW[(size_t)129 * 256 + n];
    s += 256.0f * giW[(size_t)130 * 256 + n];
    crow[n] = s;
    pooled[n] = 0.f;
  }
  __syncthreads();

  // ---- stage 1: h = lr([ce | ctrl] @ giW' + crow)  (256 x 256, K=96) ----
  // groups of 4 N-tiles: 1x A-fragment set feeds 12 WMMAs; 4 independent
  // accumulator chains hide the WMMA->VALU hazard (no v_nop stalls)
  for (int j = wv; j < 64; j += 8) {
    int rt = j >> 2, ng = (j & 3) * 4;
    v16bf a0 = *(const v16bf*)(ceFrag + ((size_t)(rt * 2 + 0) * 32 + lane) * 16);
    v16bf a1 = *(const v16bf*)(ceFrag + ((size_t)(rt * 2 + 1) * 32 + lane) * 16);
    v16bf a2;
#pragma unroll
    for (int i = 0; i < 16; ++i) a2[i] = (__bf16)0.0f;
    if (half == 0) a2[0] = ctrlcol[rt * 16 + l16];  // only k=64 nonzero
    v8f acc[4];
#pragma unroll
    for (int t = 0; t < 4; ++t) {
      float c0 = crow[(ng + t) * 16 + l16];
#pragma unroll
      for (int v = 0; v < 8; ++v) acc[t][v] = c0;
    }
#pragma unroll
    for (int t = 0; t < 4; ++t) {
      int nt = ng + t;
      v16bf b0 = *(const v16bf*)(giWs + ((size_t)(0 * 16 + nt) * 32 + lane) * 16);
      v16bf b1 = *(const v16bf*)(giWs + ((size_t)(1 * 16 + nt) * 32 + lane) * 16);
      v16bf b2 = *(const v16bf*)(giWs + ((size_t)(2 * 16 + nt) * 32 + lane) * 16);
      acc[t] = wmma_bf16(a0, b0, acc[t]);
      acc[t] = wmma_bf16(a1, b1, acc[t]);
      acc[t] = wmma_bf16(a2, b2, acc[t]);
    }
#pragma unroll
    for (int t = 0; t < 4; ++t)
#pragma unroll
      for (int v = 0; v < 8; ++v) {
        float f = LRELU(acc[t][v]);
        hbuf[(size_t)(rt * 16 + half * 8 + v) * HLD + (ng + t) * 16 + l16] = (__bf16)f;
      }
  }
  __syncthreads();

  // ---- pool h over cells (256 cols) ----
  {
    float s[8] = {0.f, 0.f, 0.f, 0.f, 0.f, 0.f, 0.f, 0.f};
    int cg = (tid & 31) * 8, rs = tid >> 5;
    for (int r = rs; r < 256; r += 8) {
      v8bf hv = *(const v8bf*)(hbuf + (size_t)r * HLD + cg);
#pragma unroll
      for (int i = 0; i < 8; ++i) s[i] += (float)hv[i];
    }
#pragma unroll
    for (int i = 0; i < 8; ++i) atomicAdd(&pooled[cg + i], s[i] * (1.0f / 256.0f));
  }
  __syncthreads();

  // pvec = pooled @ m1aW[256:512] + m1ab
  if (tid < 128) {
    float s = m1ab[tid];
    for (int jj = 0; jj < 256; ++jj)
      s += pooled[jj] * m1aW[(size_t)(256 + jj) * 128 + tid];
    pvec[tid] = s;
  }
  __syncthreads();

  // ---- m1a: a = lr(h @ m1aW[:256] + pvec)  (256 x 128, K=256) ----
  for (int j = wv; j < 64; j += 8) {
    int rt = j >> 2, n0 = (j & 3) * 2;
    v8f acc0, acc1;
    {
      float c0 = pvec[n0 * 16 + l16], c1 = pvec[(n0 + 1) * 16 + l16];
#pragma unroll
      for (int v = 0; v < 8; ++v) { acc0[v] = c0; acc1[v] = c1; }
    }
#pragma unroll
    for (int kt = 0; kt < 8; ++kt) {
      v16bf a = load_a_lds(hbuf, HLD, rt * 16 + l16, kt * 32, half);
      v16bf b0 = *(const v16bf*)(m1aWs + ((size_t)(kt * 8 + n0) * 32 + lane) * 16);
      v16bf b1 = *(const v16bf*)(m1aWs + ((size_t)(kt * 8 + n0 + 1) * 32 + lane) * 16);
      acc0 = wmma_bf16(a, b0, acc0);
      acc1 = wmma_bf16(a, b1, acc1);
    }
#pragma unroll
    for (int v = 0; v < 8; ++v) {
      int row = rt * 16 + half * 8 + v;
      abuf[(size_t)row * QLD + n0 * 16 + l16] = (__bf16)LRELU(acc0[v]);
      abuf[(size_t)row * QLD + (n0 + 1) * 16 + l16] = (__bf16)LRELU(acc1[v]);
    }
  }
  if (tid < 128) pooled[tid] = 0.f;
  __syncthreads();

  // ---- m1b: h2 = lr(a @ m1bW + m1bb)  (256 x 128, K=128) -> hbuf ld=QLD ----
  for (int j = wv; j < 64; j += 8) {
    int rt = j >> 2, n0 = (j & 3) * 2;
    v8f acc0, acc1;
    {
      float c0 = m1bb[n0 * 16 + l16], c1 = m1bb[(n0 + 1) * 16 + l16];
#pragma unroll
      for (int v = 0; v < 8; ++v) { acc0[v] = c0; acc1[v] = c1; }
    }
#pragma unroll
    for (int kt = 0; kt < 4; ++kt) {
      v16bf a = load_a_lds(abuf, QLD, rt * 16 + l16, kt * 32, half);
      v16bf b0 = *(const v16bf*)(m1bWs + ((size_t)(kt * 8 + n0) * 32 + lane) * 16);
      v16bf b1 = *(const v16bf*)(m1bWs + ((size_t)(kt * 8 + n0 + 1) * 32 + lane) * 16);
      acc0 = wmma_bf16(a, b0, acc0);
      acc1 = wmma_bf16(a, b1, acc1);
    }
#pragma unroll
    for (int v = 0; v < 8; ++v) {
      int row = rt * 16 + half * 8 + v;
      hbuf[(size_t)row * QLD + n0 * 16 + l16] = (__bf16)LRELU(acc0[v]);
      hbuf[(size_t)row * QLD + (n0 + 1) * 16 + l16] = (__bf16)LRELU(acc1[v]);
    }
  }
  __syncthreads();

  // ---- pool h2 (128 cols) ----
  {
    float s[8] = {0.f, 0.f, 0.f, 0.f, 0.f, 0.f, 0.f, 0.f};
    int cg = (tid & 15) * 8, rs = tid >> 4;
    for (int r = rs; r < 256; r += 16) {
      v8bf hv = *(const v8bf*)(hbuf + (size_t)r * QLD + cg);
#pragma unroll
      for (int i = 0; i < 8; ++i) s[i] += (float)hv[i];
    }
#pragma unroll
    for (int i = 0; i < 8; ++i) atomicAdd(&pooled[cg + i], s[i] * (1.0f / 256.0f));
  }
  __syncthreads();

  if (tid < 128) {
    float s = m2ab[tid];
    for (int jj = 0; jj < 128; ++jj)
      s += pooled[jj] * m2aW[(size_t)(128 + jj) * 128 + tid];
    pvec[tid] = s;
  }
  __syncthreads();

  // ---- m2a: a2 = lr(h2 @ m2aW[:128] + pvec)  K=128 -> abuf ----
  for (int j = wv; j < 64; j += 8) {
    int rt = j >> 2, n0 = (j & 3) * 2;
    v8f acc0, acc1;
    {
      float c0 = pvec[n0 * 16 + l16], c1 = pvec[(n0 + 1) * 16 + l16];
#pragma unroll
      for (int v = 0; v < 8; ++v) { acc0[v] = c0; acc1[v] = c1; }
    }
#pragma unroll
    for (int kt = 0; kt < 4; ++kt) {
      v16bf a = load_a_lds(hbuf, QLD, rt * 16 + l16, kt * 32, half);
      v16bf b0 = *(const v16bf*)(m2aWs + ((size_t)(kt * 8 + n0) * 32 + lane) * 16);
      v16bf b1 = *(const v16bf*)(m2aWs + ((size_t)(kt * 8 + n0 + 1) * 32 + lane) * 16);
      acc0 = wmma_bf16(a, b0, acc0);
      acc1 = wmma_bf16(a, b1, acc1);
    }
#pragma unroll
    for (int v = 0; v < 8; ++v) {
      int row = rt * 16 + half * 8 + v;
      abuf[(size_t)row * QLD + n0 * 16 + l16] = (__bf16)LRELU(acc0[v]);
      abuf[(size_t)row * QLD + (n0 + 1) * 16 + l16] = (__bf16)LRELU(acc1[v]);
    }
  }
  __syncthreads();

  // ---- m2b: h3 = lr(a2 @ m2bW + m2bb)  K=128 -> hbuf ld=QLD ----
  for (int j = wv; j < 64; j += 8) {
    int rt = j >> 2, n0 = (j & 3) * 2;
    v8f acc0, acc1;
    {
      float c0 = m2bb[n0 * 16 + l16], c1 = m2bb[(n0 + 1) * 16 + l16];
#pragma unroll
      for (int v = 0; v < 8; ++v) { acc0[v] = c0; acc1[v] = c1; }
    }
#pragma unroll
    for (int kt = 0; kt < 4; ++kt) {
      v16bf a = load_a_lds(abuf, QLD, rt * 16 + l16, kt * 32, half);
      v16bf b0 = *(const v16bf*)(m2bWs + ((size_t)(kt * 8 + n0) * 32 + lane) * 16);
      v16bf b1 = *(const v16bf*)(m2bWs + ((size_t)(kt * 8 + n0 + 1) * 32 + lane) * 16);
      acc0 = wmma_bf16(a, b0, acc0);
      acc1 = wmma_bf16(a, b1, acc1);
    }
#pragma unroll
    for (int v = 0; v < 8; ++v) {
      int row = rt * 16 + half * 8 + v;
      hbuf[(size_t)row * QLD + n0 * 16 + l16] = (__bf16)LRELU(acc0[v]);
      hbuf[(size_t)row * QLD + (n0 + 1) * 16 + l16] = (__bf16)LRELU(acc1[v]);
    }
  }
  __syncthreads();

  // ---- output head + softmax over the 256 cells of this gene ----
  float vo = ob[0];
  {
    const __bf16* rowp = hbuf + (size_t)tid * QLD;
#pragma unroll 4
    for (int k = 0; k < 128; k += 8) {
      v8bf hv = *(const v8bf*)(rowp + k);
#pragma unroll
      for (int i = 0; i < 8; ++i) vo += (float)hv[i] * oW[k + i];
    }
  }
  float mx = vo;
#pragma unroll
  for (int off = 16; off > 0; off >>= 1) mx = fmaxf(mx, __shfl_xor(mx, off, 32));
  if (lane == 0) redm[wv] = mx;
  __syncthreads();
  if (tid == 0) {
    float m = redm[0];
    for (int i = 1; i < 8; ++i) m = fmaxf(m, redm[i]);
    redm[0] = m;
  }
  __syncthreads();
  float e = __expf(vo - redm[0]);
  float sm = e;
#pragma unroll
  for (int off = 16; off > 0; off >>= 1) sm += __shfl_xor(sm, off, 32);
  if (lane == 0) reds[wv] = sm;
  __syncthreads();
  if (tid == 0) {
    float s = reds[0];
    for (int i = 1; i < 8; ++i) s += reds[i];
    reds[0] = s;
  }
  __syncthreads();
  out[(size_t)tid * NGENES + g] = e / reds[0];
}

// ---------------- host launcher ----------------

extern "C" void kernel_launch(void* const* d_in, const int* in_sizes, int n_in,
                              void* d_out, int out_size, void* d_ws, size_t ws_size,
                              hipStream_t stream) {
  const float* ctrl       = (const float*)d_in[0];
  const float* mean_shift = (const float*)d_in[1];
  const float* cw0 = (const float*)d_in[2];  const float* cb0 = (const float*)d_in[3];
  const float* cw1 = (const float*)d_in[4];  const float* cb1 = (const float*)d_in[5];
  const float* cw2 = (const float*)d_in[6];  const float* cb2 = (const float*)d_in[7];
  const float* cw3 = (const float*)d_in[8];  const float* cb3 = (const float*)d_in[9];
  const float* gene_table = (const float*)d_in[10];
  const float* giW  = (const float*)d_in[11]; const float* gib  = (const float*)d_in[12];
  const float* m1aW = (const float*)d_in[13]; const float* m1ab = (const float*)d_in[14];
  const float* m1bW = (const float*)d_in[15]; const float* m1bb = (const float*)d_in[16];
  const float* m2aW = (const float*)d_in[17]; const float* m2ab = (const float*)d_in[18];
  const float* m2bW = (const float*)d_in[19]; const float* m2bb = (const float*)d_in[20];
  const float* oW   = (const float*)d_in[21]; const float* ob   = (const float*)d_in[22];
  float* out = (float*)d_out;

  // workspace layout
  char* w = (char*)d_ws;
  float* ce0 = (float*)w;            w += (size_t)256 * 128 * 4;
  float* ce1 = (float*)w;            w += (size_t)256 * 128 * 4;
  float* ce2 = (float*)w;            w += (size_t)256 * 128 * 4;
  float* ce  = (float*)w;            w += (size_t)256 * 64 * 4;
  __bf16* ceFrag = (__bf16*)w;       w += (size_t)16 * 2 * 32 * 16 * 2;
  __bf16* giWs   = (__bf16*)w;       w += (size_t)3 * 16 * 512 * 2;
  __bf16* m1aWs  = (__bf16*)w;       w += (size_t)8 * 8 * 512 * 2;
  __bf16* m1bWs  = (__bf16*)w;       w += (size_t)4 * 8 * 512 * 2;
  __bf16* m2aWs  = (__bf16*)w;       w += (size_t)4 * 8 * 512 * 2;
  __bf16* m2bWs  = (__bf16*)w;       w += (size_t)4 * 8 * 512 * 2;

  // cell embedder MLP (gene-independent, tiny)
  mlp_layer<<<(256 * 128 + 255) / 256, 256, 0, stream>>>(ctrl, cw0, cb0, ce0, 256, 2000, 128, 1);
  mlp_layer<<<(256 * 128 + 255) / 256, 256, 0, stream>>>(ce0, cw1, cb1, ce1, 256, 128, 128, 1);
  mlp_layer<<<(256 * 128 + 255) / 256, 256, 0, stream>>>(ce1, cw2, cb2, ce2, 256, 128, 128, 1);
  mlp_layer<<<(256 * 64 + 255) / 256, 256, 0, stream>>>(ce2, cw3, cb3, ce, 256, 128, 64, 0);

  // fragment pre-swizzles
  build_ce_frags<<<4, 256, 0, stream>>>(ce, ceFrag);
  swizzle_giw<<<6, 256, 0, stream>>>(giW, giWs);
  swizzle_b<<<(8 * 8 * 32 + 255) / 256, 256, 0, stream>>>(m1aW, m1aWs, 256, 128);
  swizzle_b<<<(4 * 8 * 32 + 255) / 256, 256, 0, stream>>>(m1bW, m1bWs, 128, 128);
  swizzle_b<<<(4 * 8 * 32 + 255) / 256, 256, 0, stream>>>(m2aW, m2aWs, 128, 128);
  swizzle_b<<<(4 * 8 * 32 + 255) / 256, 256, 0, stream>>>(m2bW, m2bWs, 128, 128);

  // fused per-gene pipeline
  fcgnn_gene_kernel<<<NGENES, 256, 0, stream>>>(
      ctrl, mean_shift, gene_table, giW, gib,
      m1aW, m1ab, m1bb, m2aW, m2ab, m2bb, oW, ob,
      ceFrag, giWs, m1aWs, m1bWs, m2aWs, m2bWs, out);
}
